// GSALayer_74646531604560
// MI455X (gfx1250) — matrix operand order repaired and checked by reading, not compiled, not run on hardware
//
#include <hip/hip_runtime.h>
#include <hip/hip_bf16.h>

// GSA layer for MI455X (gfx1250, wave32, WMMA).
// B=4, N=4096 -> BN=16384 tokens; H=4096; D=256; M=64 slots.

typedef __attribute__((ext_vector_type(16))) _Float16 v16h;
typedef __attribute__((ext_vector_type(8)))  _Float16 v8h;
typedef __attribute__((ext_vector_type(8)))  float    v8f;
typedef __attribute__((ext_vector_type(4)))  int      v4i;

#define HID  4096
#define DSL  256
#define MSL  64
#define NTOK 4096
#define BN   16384
#define BSTR 80    // LDS B row stride (halves) for 64-wide tiles
#define PW   192   // projection GEMM: cols per WG
#define PBS  208   // LDS B row stride for 192-wide tiles  (208*2 % 32 == 0)
#define OW   128   // output GEMM: cols per WG
#define OBS  144   // LDS B row stride for 128-wide tiles  (144*2 % 32 == 0)

// ---- async copy (CDNA5 GLOBAL_LOAD_ASYNC_TO_LDS), guarded ---------------

#if defined(__has_builtin)
#if __has_builtin(__builtin_amdgcn_global_load_async_to_lds_b128)
#define HAVE_ASYNC 1
#endif
#endif
#ifndef HAVE_ASYNC
#define HAVE_ASYNC 0
#endif

#define AS1 __attribute__((address_space(1)))
#define AS3 __attribute__((address_space(3)))

__device__ __forceinline__ void async_b128(const _Float16* g, _Float16* l) {
#if HAVE_ASYNC
    // builtin signature: (v4i AS1* src, v4i AS3* dst, imm offset, imm cpol)
    __builtin_amdgcn_global_load_async_to_lds_b128((AS1 v4i*)g, (AS3 v4i*)l, 0, 0);
#else
    *(v8h*)l = *(const v8h*)g;   // sync fallback through VGPRs
#endif
}
__device__ __forceinline__ void async_wait() {
#if HAVE_ASYNC
#if __has_builtin(__builtin_amdgcn_s_wait_asynccnt)
    __builtin_amdgcn_s_wait_asynccnt(0);
#else
    asm volatile("s_wait_asynccnt 0x0" ::: "memory");
#endif
#endif
}

// ---- helpers -------------------------------------------------------------

__device__ __forceinline__ v16h load_cvt16(const float* p) {
    const float4* q = (const float4*)p;
    float4 a = q[0], b = q[1], c = q[2], d = q[3];
    v16h h;
    h[0]  = (_Float16)a.x; h[1]  = (_Float16)a.y; h[2]  = (_Float16)a.z; h[3]  = (_Float16)a.w;
    h[4]  = (_Float16)b.x; h[5]  = (_Float16)b.y; h[6]  = (_Float16)b.z; h[7]  = (_Float16)b.w;
    h[8]  = (_Float16)c.x; h[9]  = (_Float16)c.y; h[10] = (_Float16)c.z; h[11] = (_Float16)c.w;
    h[12] = (_Float16)d.x; h[13] = (_Float16)d.y; h[14] = (_Float16)d.z; h[15] = (_Float16)d.w;
    return h;
}

// 64x32 f32 tile -> f16 LDS A (row stride 32 halves). 128 threads.
__device__ __forceinline__ void load_a_tile_f32(_Float16* As, const float* src, int ld, int tid) {
    int row = tid >> 1, cb = (tid & 1) << 4;
    *(v16h*)(As + row * 32 + cb) = load_cvt16(src + (size_t)row * ld + cb);
}
// 64x32 f16 tile -> LDS A via async copy (2 x b128 per thread).
__device__ __forceinline__ void load_a_tile_f16_async(_Float16* As, const _Float16* src, int ld, int tid) {
#pragma unroll
    for (int c = 0; c < 2; ++c) {
        int idx = tid * 2 + c;
        int row = idx >> 2, col = (idx & 3) * 8;
        async_b128(src + (size_t)row * ld + col, As + row * 32 + col);
    }
}
// 32x64 f16 tile -> LDS B (sync, for small kernels).
__device__ __forceinline__ void load_b_tile_f16(_Float16* Bs, const _Float16* src, int ld, int tid) {
    int row = tid >> 2, cb = (tid & 3) << 4;
    *(v16h*)(Bs + row * BSTR + cb) = *(const v16h*)(src + (size_t)row * ld + cb);
}
// 32x64 f32 tile -> f16 LDS B.
__device__ __forceinline__ void load_b_tile_f32(_Float16* Bs, const float* src, int ld, int tid) {
    int row = tid >> 2, cb = (tid & 3) << 4;
    *(v16h*)(Bs + row * BSTR + cb) = load_cvt16(src + (size_t)row * ld + cb);
}
// 32 x W f16 tile -> LDS B via async b128 copies. W multiple of 64.
template<int W, int BS>
__device__ __forceinline__ void load_b_async(_Float16* Bs, const _Float16* src, int ld, int tid) {
    constexpr int CPT = (32 * W / 8) / 128;   // b128 chunks per thread
#pragma unroll
    for (int c = 0; c < CPT; ++c) {
        int idx = tid * CPT + c;
        int row = idx / (W / 8);
        int col = (idx % (W / 8)) * 8;
        async_b128(src + (size_t)row * ld + col, Bs + row * BS + col);
    }
}

// A fragment (16x32 f16): lane L<16: row M=L, K {0..7,16..23}; L>=16: K {8..15,24..31}.
__device__ __forceinline__ v16h a_frag(const _Float16* As, int lane, int wrow) {
    int row = wrow + (lane & 15);
    int kh  = (lane >> 4) << 3;
    v8h lo = *(const v8h*)(As + row * 32 + kh);
    v8h hi = *(const v8h*)(As + row * 32 + kh + 16);
    return __builtin_shufflevector(lo, hi, 0,1,2,3,4,5,6,7,8,9,10,11,12,13,14,15);
}

// Each wave: 16-row strip x NT*16 cols.
template<int NT, int BS>
__device__ __forceinline__ void wmmaN(const _Float16* As, const _Float16* Bs,
                                      int lane, int wrow, v8f* acc) {
    v16h a = a_frag(As, lane, wrow);
#pragma unroll
    for (int t = 0; t < NT; ++t) {
        v16h b;
        int bn = t * 16 + (lane & 15), bkh = (lane >> 4) << 3;
#pragma unroll
        for (int s = 0; s < 16; ++s)
            b[s] = Bs[(bkh + (s & 7) + ((s >> 3) << 4)) * BS + bn];
        acc[t] = __builtin_amdgcn_wmma_f32_16x16x32_f16(
            false, a, false, b, (short)0, acc[t], false, false);
    }
}

// ---- kernel 1: pack weights (f32 [out,in] -> f16 [K][N]) -----------------

__global__ void gsa_pack(const float* __restrict__ qw, const float* __restrict__ vw,
                         const float* __restrict__ gw, const float* __restrict__ ow,
                         const float* __restrict__ slots,
                         _Float16* __restrict__ wcatT, _Float16* __restrict__ owT,
                         _Float16* __restrict__ slotsT) {
    size_t i = (size_t)blockIdx.x * blockDim.x + threadIdx.x;
    const size_t N1 = (size_t)HID * 576;
    const size_t N2 = (size_t)DSL * HID;
    const size_t N3 = (size_t)DSL * MSL;
    if (i < N1) {
        size_t k = i / 576; int n = (int)(i % 576);
        float val = (n < 256) ? qw[(size_t)n * HID + k]
                  : (n < 512) ? vw[(size_t)(n - 256) * HID + k]
                              : gw[(size_t)(n - 512) * HID + k];
        wcatT[i] = (_Float16)val;
    } else if (i < N1 + N2) {
        size_t j = i - N1; size_t k = j / HID; int n = (int)(j % HID);
        owT[j] = (_Float16)ow[(size_t)n * DSL + k];
    } else if (i < N1 + N2 + N3) {
        size_t j = i - N1 - N2; size_t k = j / MSL; int m = (int)(j % MSL);
        slotsT[j] = (_Float16)slots[(size_t)m * DSL + k];
    }
}

// ---- kernel 2: fused q/v/gate projection (double-buffered, 192-wide) -----

__global__ void __launch_bounds__(128) gsa_proj(
        const float* __restrict__ hs, const _Float16* __restrict__ wcatT,
        const float* __restrict__ qb, const float* __restrict__ vb,
        const float* __restrict__ gb,
        float* __restrict__ qo, float* __restrict__ vo, float* __restrict__ go) {
    __shared__ __align__(32) _Float16 As[2][64 * 32];
    __shared__ __align__(32) _Float16 Bs[2][32 * PBS];
    int tid = threadIdx.x, lane = tid & 31, w = tid >> 5;
    int ncb = blockIdx.x * PW;                 // 0, 192, 384
    size_t rowbase = (size_t)blockIdx.y * 64;
    v8f acc[12] = {};
    // prologue: chunk 0 -> buffer 0
    load_a_tile_f32(As[0], hs + rowbase * HID, HID, tid);
    load_b_async<PW, PBS>(Bs[0], wcatT + ncb, 576, tid);
    async_wait();
    __syncthreads();
    for (int kb = 0; kb < HID; kb += 32) {
        int cur = (kb >> 5) & 1;
        if (kb + 32 < HID) {
            load_a_tile_f32(As[cur ^ 1], hs + rowbase * HID + kb + 32, HID, tid);
            load_b_async<PW, PBS>(Bs[cur ^ 1], wcatT + (size_t)(kb + 32) * 576 + ncb, 576, tid);
            __builtin_prefetch((const void*)(hs + (rowbase + (tid >> 1)) * HID + kb + 64), 0, 3);
        }
        wmmaN<12, PBS>(As[cur], Bs[cur], lane, w * 16, acc);
        async_wait();
        __syncthreads();
    }
    int rl = (lane >> 4) << 3, cl = lane & 15;
#pragma unroll
    for (int t = 0; t < 12; ++t) {
        int c0 = ncb + t * 16;   // 16-col tile lies fully in one region
        float* dst; const float* bias; int ldd, coff;
        if (c0 < 256)      { dst = qo; bias = qb; ldd = 256; coff = 0;   }
        else if (c0 < 512) { dst = vo; bias = vb; ldd = 256; coff = 256; }
        else               { dst = go; bias = gb; ldd = 64;  coff = 512; }
        int gc = c0 + cl - coff;
#pragma unroll
        for (int r = 0; r < 8; ++r) {
            size_t gr = rowbase + w * 16 + r + rl;
            dst[gr * ldd + gc] = acc[t][r] + bias[gc];
        }
    }
}

// ---- kernel 3: slot weights = softmax(q @ slots^T), masked ---------------

__global__ void __launch_bounds__(128) gsa_slotw(
        const float* __restrict__ q, const _Float16* __restrict__ slotsT,
        const unsigned char* __restrict__ amask, float* __restrict__ sw) {
    __shared__ __align__(32) _Float16 As[64 * 32];
    __shared__ __align__(32) _Float16 Bs[32 * BSTR];
    __shared__ float S[64 * 65];
    int tid = threadIdx.x, lane = tid & 31, w = tid >> 5;
    size_t rowbase = (size_t)blockIdx.x * 64;
    v8f acc[4] = {};
    for (int kb = 0; kb < DSL; kb += 32) {
        load_a_tile_f32(As, q + rowbase * DSL + kb, DSL, tid);
        load_b_tile_f16(Bs, slotsT + (size_t)kb * MSL, MSL, tid);
        __syncthreads();
        wmmaN<4, BSTR>(As, Bs, lane, w * 16, acc);
        __syncthreads();
    }
    int rl = (lane >> 4) << 3, cl = lane & 15;
#pragma unroll
    for (int t = 0; t < 4; ++t)
#pragma unroll
        for (int r = 0; r < 8; ++r)
            S[(w * 16 + r + rl) * 65 + t * 16 + cl] = acc[t][r];
    __syncthreads();
    if (tid < 64) {
        size_t row = rowbase + tid;
        bool ok = amask[row] != 0;
        float* sr = S + tid * 65;
        float m = -1e30f;
        for (int j = 0; j < 64; ++j) m = fmaxf(m, sr[j]);
        float ssum = 0.f;
        for (int j = 0; j < 64; ++j) { float e = __expf(sr[j] - m); sr[j] = e; ssum += e; }
        float inv = ok ? (1.0f / ssum) : 0.0f;
        for (int j = 0; j < 64; ++j) sw[row * 64 + j] = sr[j] * inv;
    }
}

// ---- kernel 4: gates = softmax(gate logits) ------------------------------

__global__ void gsa_gates(const float* __restrict__ g, float* __restrict__ gout) {
    size_t row = (size_t)blockIdx.x * blockDim.x + threadIdx.x;
    if (row >= BN) return;
    const float* gr = g + row * 64;
    float m = -1e30f;
    for (int j = 0; j < 64; ++j) m = fmaxf(m, gr[j]);
    float s = 0.f;
    for (int j = 0; j < 64; ++j) s += __expf(gr[j] - m);
    float inv = 1.0f / s;
    for (int j = 0; j < 64; ++j) gout[row * 64 + j] = __expf(gr[j] - m) * inv;
}

// ---- kernel 5: slot_values[b] = sw[b]^T @ v[b] ---------------------------

__global__ void __launch_bounds__(128) gsa_slotvals(
        const float* __restrict__ sw, const float* __restrict__ v,
        _Float16* __restrict__ sv) {
    __shared__ __align__(32) _Float16 As[64 * 32];
    __shared__ __align__(32) _Float16 Bs[32 * BSTR];
    int tid = threadIdx.x, lane = tid & 31, w = tid >> 5;
    int b = blockIdx.y;
    int dcb = blockIdx.x * 64;
    const float* swb = sw + (size_t)b * NTOK * MSL;
    const float* vpb = v  + (size_t)b * NTOK * DSL;
    v8f acc[4] = {};
    for (int kb = 0; kb < NTOK; kb += 32) {
        {   // transposed A load: As[m][i] = sw[kb+i][m] (f32 -> f16)
            int i = tid >> 2; int mb = (tid & 3) << 4;
            v16h h = load_cvt16(swb + (size_t)(kb + i) * MSL + mb);
#pragma unroll
            for (int j = 0; j < 16; ++j) As[(mb + j) * 32 + i] = h[j];
        }
        load_b_tile_f32(Bs, vpb + (size_t)kb * DSL + dcb, DSL, tid);
        __syncthreads();
        wmmaN<4, BSTR>(As, Bs, lane, w * 16, acc);
        __syncthreads();
    }
    int rl = (lane >> 4) << 3, cl = lane & 15;
#pragma unroll
    for (int t = 0; t < 4; ++t)
#pragma unroll
        for (int r = 0; r < 8; ++r) {
            int m = w * 16 + r + rl, d = dcb + t * 16 + cl;
            sv[((size_t)b * MSL + m) * DSL + d] = (_Float16)acc[t][r];
        }
}

// ---- kernel 6: gated_values = gates @ slot_values ------------------------

__global__ void __launch_bounds__(128) gsa_gated(
        const float* __restrict__ gates, const _Float16* __restrict__ sv,
        _Float16* __restrict__ gv) {
    __shared__ __align__(32) _Float16 As[64 * 32];
    __shared__ __align__(32) _Float16 Bs[32 * BSTR];
    int tid = threadIdx.x, lane = tid & 31, w = tid >> 5;
    size_t rowbase = (size_t)blockIdx.y * 64;
    int b = (int)(rowbase >> 12);
    int dcb = blockIdx.x * 64;
    v8f acc[4] = {};
    for (int kb = 0; kb < MSL; kb += 32) {
        load_a_tile_f32(As, gates + rowbase * MSL + kb, MSL, tid);
        load_b_tile_f16(Bs, sv + ((size_t)b * MSL + kb) * DSL + dcb, DSL, tid);
        __syncthreads();
        wmmaN<4, BSTR>(As, Bs, lane, w * 16, acc);
        __syncthreads();
    }
    int rl = (lane >> 4) << 3, cl = lane & 15;
#pragma unroll
    for (int t = 0; t < 4; ++t)
#pragma unroll
        for (int r = 0; r < 8; ++r)
            gv[(rowbase + w * 16 + r + rl) * DSL + dcb + t * 16 + cl] = (_Float16)acc[t][r];
}

// ---- kernel 7: output = gated_values @ o_w^T + o_b (double-buffered) -----

__global__ void __launch_bounds__(128) gsa_out(
        const _Float16* __restrict__ gv, const _Float16* __restrict__ owT,
        const float* __restrict__ ob, float* __restrict__ out) {
    __shared__ __align__(32) _Float16 As[2][64 * 32];
    __shared__ __align__(32) _Float16 Bs[2][32 * OBS];
    int tid = threadIdx.x, lane = tid & 31, w = tid >> 5;
    size_t rowbase = (size_t)blockIdx.y * 64;
    int ncb = blockIdx.x * OW;
    v8f acc[8] = {};
    load_a_tile_f16_async(As[0], gv + rowbase * DSL, DSL, tid);
    load_b_async<OW, OBS>(Bs[0], owT + ncb, HID, tid);
    async_wait();
    __syncthreads();
    for (int kb = 0; kb < DSL; kb += 32) {
        int cur = (kb >> 5) & 1;
        if (kb + 32 < DSL) {
            load_a_tile_f16_async(As[cur ^ 1], gv + rowbase * DSL + kb + 32, DSL, tid);
            load_b_async<OW, OBS>(Bs[cur ^ 1], owT + (size_t)(kb + 32) * HID + ncb, HID, tid);
        }
        wmmaN<8, OBS>(As[cur], Bs[cur], lane, w * 16, acc);
        async_wait();
        __syncthreads();
    }
    int rl = (lane >> 4) << 3, cl = lane & 15;
#pragma unroll
    for (int t = 0; t < 8; ++t)
#pragma unroll
        for (int r = 0; r < 8; ++r) {
            size_t gr = rowbase + w * 16 + r + rl;
            int gc = ncb + t * 16 + cl;
            out[gr * HID + gc] = acc[t][r] + ob[gc];
        }
}

// ---- launch --------------------------------------------------------------

extern "C" void kernel_launch(void* const* d_in, const int* in_sizes, int n_in,
                              void* d_out, int out_size, void* d_ws, size_t ws_size,
                              hipStream_t stream) {
    (void)in_sizes; (void)n_in; (void)out_size; (void)ws_size;
    const float*         hs    = (const float*)d_in[0];
    const unsigned char* amask = (const unsigned char*)d_in[1];
    const float*         slots = (const float*)d_in[2];
    const float*         q_w   = (const float*)d_in[3];
    const float*         q_b   = (const float*)d_in[4];
    // d_in[5]=k_w, d_in[6]=k_b are dead in the reference -> skipped
    const float*         v_w   = (const float*)d_in[7];
    const float*         v_b   = (const float*)d_in[8];
    const float*         g_w   = (const float*)d_in[9];
    const float*         g_b   = (const float*)d_in[10];
    const float*         o_w   = (const float*)d_in[11];
    const float*         o_b   = (const float*)d_in[12];

    float* outp      = (float*)d_out;              // [BN, HID]
    float* gates_out = outp + (size_t)BN * HID;    // [BN, MSL]

    char* wsb = (char*)d_ws;
    size_t off = 0;
    auto take = [&](size_t bytes) -> void* {
        void* p = wsb + off; off += (bytes + 255) & ~(size_t)255; return p;
    };
    _Float16* wcatT  = (_Float16*)take((size_t)HID * 576 * 2);
    _Float16* owT    = (_Float16*)take((size_t)DSL * HID * 2);
    _Float16* slotsT = (_Float16*)take((size_t)DSL * MSL * 2);
    float*    qbuf   = (float*)   take((size_t)BN * DSL * 4);
    float*    vbuf   = (float*)   take((size_t)BN * DSL * 4);
    float*    gbuf   = (float*)   take((size_t)BN * MSL * 4);
    float*    swbuf  = (float*)   take((size_t)BN * MSL * 4);
    _Float16* svbuf  = (_Float16*)take((size_t)4 * MSL * DSL * 2);
    _Float16* gvbuf  = (_Float16*)take((size_t)BN * DSL * 2);

    size_t packN = (size_t)HID * 576 + (size_t)DSL * HID + (size_t)DSL * MSL;
    gsa_pack<<<(unsigned)((packN + 255) / 256), 256, 0, stream>>>(
        q_w, v_w, g_w, o_w, slots, wcatT, owT, slotsT);
    gsa_proj<<<dim3(576 / PW, BN / 64), 128, 0, stream>>>(
        hs, wcatT, q_b, v_b, g_b, qbuf, vbuf, gbuf);
    gsa_slotw<<<BN / 64, 128, 0, stream>>>(qbuf, slotsT, amask, swbuf);
    gsa_gates<<<BN / 128, 128, 0, stream>>>(gbuf, gates_out);
    gsa_slotvals<<<dim3(4, 4), 128, 0, stream>>>(swbuf, vbuf, svbuf);
    gsa_gated<<<dim3(4, BN / 64), 128, 0, stream>>>(gates_out, svbuf, gvbuf);
    gsa_out<<<dim3(HID / OW, BN / 64), 128, 0, stream>>>(gvbuf, owT, o_b, outp);
}
